// CharGRU_57836029608477
// MI455X (gfx1250) — matrix operand order repaired and checked by reading, not compiled, not run on hardware
//
#include <hip/hip_runtime.h>
#include <hip/hip_bf16.h>
#include <math.h>

// ---- CDNA5 WMMA types (wave32) ----
typedef __attribute__((ext_vector_type(16))) __bf16 v16bf;
typedef __attribute__((ext_vector_type(8)))  float  v8f;

#define HID   1024
#define EMB   256
#define BAT   64
#define SEQL  512
#define VOC   128

// round-to-nearest-even f32 -> bf16 (raw ushort)
__device__ __forceinline__ unsigned short f2bf(float f) {
  unsigned int u = __float_as_uint(f);
  u += 0x7FFFu + ((u >> 16) & 1u);
  return (unsigned short)(u >> 16);
}

union Frag { v16bf v; uint4 u[2]; };

// A-operand fragment: 16x32 bf16 tile from row-major matrix (row stride ld).
// Layout: lanes 0-15 hold M=lane, K=kc+[0..15]; lanes 16-31 hold M=lane-16,
// K=kc+[16..31] (matches load_b_frag k order).
__device__ __forceinline__ v16bf load_a_frag(const unsigned short* __restrict__ base,
                                             int ld, int mrow, int kc, int lane) {
  const int l15 = lane & 15, hi = lane >> 4;
  const unsigned short* p = base + (size_t)(mrow + l15) * ld + kc + hi * 16;
  Frag f;
  f.u[0] = *(const uint4*)(p);        // 8 halves, 16B aligned
  f.u[1] = *(const uint4*)(p + 8);    // next 8 K (contiguous)
  return f.v;
}

// B-operand fragment: 32x16 tile with B[k][n] = W[nb+n][kc+k], W row-major (ld).
// Layout (contiguous k, same as A): lanes 0-15 hold N=lane, K=kc+[0..15]; lanes 16-31 hold N=lane-16,
// K=kc+[16..31]. One contiguous 32B row segment per lane.
__device__ __forceinline__ v16bf load_b_frag(const unsigned short* __restrict__ base,
                                             int ld, int nb, int kc, int lane) {
  const int l15 = lane & 15, hi = lane >> 4;
  const unsigned short* p = base + (size_t)(nb + l15) * ld + kc + hi * 16;
  Frag f;
  f.u[0] = *(const uint4*)(p);
  f.u[1] = *(const uint4*)(p + 8);
  return f.v;
}

__device__ __forceinline__ v8f wmma_bf16(v16bf a, v16bf b, v8f c) {
  // (neg_a, A, neg_b, B, c_mod, C, reuse_a, reuse_b)
  return __builtin_amdgcn_wmma_f32_16x16x32_bf16(false, a, false, b, (short)0, c,
                                                 false, false);
}

// ---------------- prep kernels ----------------
__global__ void k_f32_to_bf16(const float* __restrict__ s,
                              unsigned short* __restrict__ d, int n) {
  int i = blockIdx.x * blockDim.x + threadIdx.x;
  if (i < n) d[i] = f2bf(s[i]);
}

// X[s][b][e] = bf16(emb[x[b][s]][e]); grid covers SEQL*BAT*EMB exactly
__global__ void k_embed(const int* __restrict__ x, const float* __restrict__ emb,
                        unsigned short* __restrict__ X) {
  int i = blockIdx.x * blockDim.x + threadIdx.x;
  int e = i & (EMB - 1);
  int b = (i >> 8) & (BAT - 1);
  int s = i >> 14;                 // / (BAT*EMB)
  int tok = x[b * SEQL + s];
  X[i] = f2bf(emb[tok * EMB + e]);
}

__global__ void k_init_h(float* __restrict__ hf, unsigned short* __restrict__ hb) {
  int i = blockIdx.x * blockDim.x + threadIdx.x;   // BAT*HID exactly
  hf[i] = 0.0f;
  hb[i] = 0;
}

// ---------------- per-step kernels ----------------
// r/z gates: 256 waves, each owns one 16x16 tile of [64 x 1024].
__global__ void k_gates(const unsigned short* __restrict__ X,
                        const unsigned short* __restrict__ Wr,
                        const unsigned short* __restrict__ Wz,
                        const unsigned short* __restrict__ Ur,
                        const unsigned short* __restrict__ Uz,
                        const unsigned short* __restrict__ hb,
                        const float* __restrict__ hf,
                        const float* __restrict__ br,
                        const float* __restrict__ bz,
                        float* __restrict__ zbuf,
                        unsigned short* __restrict__ rh, int s) {
  const int lane = threadIdx.x & 31;
  const int wid  = blockIdx.x * (blockDim.x >> 5) + (threadIdx.x >> 5);
  const int mt = wid & 3, nt = wid >> 2;          // 4 x 64 tiles
  const int mrow = mt * 16, nb = nt * 16;
  const unsigned short* Xs = X + (size_t)s * BAT * EMB;
  v8f ar = {}, az = {};
  // input projection: K = EMB (A fragment shared across both gates)
  #pragma unroll
  for (int kb = 0; kb < EMB / 32; ++kb) {
    const int kc = kb * 32;
    v16bf a = load_a_frag(Xs, EMB, mrow, kc, lane);
    ar = wmma_bf16(a, load_b_frag(Wr, EMB, nb, kc, lane), ar);
    az = wmma_bf16(a, load_b_frag(Wz, EMB, nb, kc, lane), az);
  }
  // recurrent: K = HID
  #pragma unroll 4
  for (int kb = 0; kb < HID / 32; ++kb) {
    const int kc = kb * 32;
    v16bf a = load_a_frag(hb, HID, mrow, kc, lane);
    ar = wmma_bf16(a, load_b_frag(Ur, HID, nb, kc, lane), ar);
    az = wmma_bf16(a, load_b_frag(Uz, HID, nb, kc, lane), az);
  }
  // epilogue: D layout — lane holds N = nb+(lane&15); VGPR v holds M = mrow+v(+8)
  const int l15 = lane & 15, hi = lane >> 4;
  const int n = nb + l15;
  const float tbr = 2.0f * br[n], tbz = 2.0f * bz[n];  // reference adds bias twice
  #pragma unroll
  for (int v = 0; v < 8; ++v) {
    const int m = mrow + v + hi * 8;
    const float r = 1.0f / (1.0f + expf(-(ar[v] + tbr)));
    const float z = 1.0f / (1.0f + expf(-(az[v] + tbz)));
    zbuf[m * HID + n] = z;
    rh[m * HID + n]   = f2bf(r * hf[m * HID + n]);
  }
}

// candidate + state update: h = (1-z)*tanh(xh + (r*h)U_h^T + 2b) + z*h
__global__ void k_cand(const unsigned short* __restrict__ X,
                       const unsigned short* __restrict__ Wh,
                       const unsigned short* __restrict__ Uh,
                       const unsigned short* __restrict__ rh,
                       const float* __restrict__ bh,
                       const float* __restrict__ zbuf,
                       float* __restrict__ hf,
                       unsigned short* __restrict__ hb,
                       float* __restrict__ hlast, int s, int last) {
  const int lane = threadIdx.x & 31;
  const int wid  = blockIdx.x * (blockDim.x >> 5) + (threadIdx.x >> 5);
  const int mt = wid & 3, nt = wid >> 2;
  const int mrow = mt * 16, nb = nt * 16;
  const unsigned short* Xs = X + (size_t)s * BAT * EMB;
  v8f acc = {};
  #pragma unroll
  for (int kb = 0; kb < EMB / 32; ++kb) {
    const int kc = kb * 32;
    acc = wmma_bf16(load_a_frag(Xs, EMB, mrow, kc, lane),
                    load_b_frag(Wh, EMB, nb, kc, lane), acc);
  }
  #pragma unroll 4
  for (int kb = 0; kb < HID / 32; ++kb) {
    const int kc = kb * 32;
    acc = wmma_bf16(load_a_frag(rh, HID, mrow, kc, lane),
                    load_b_frag(Uh, HID, nb, kc, lane), acc);
  }
  const int l15 = lane & 15, hi = lane >> 4;
  const int n = nb + l15;
  const float tbh = 2.0f * bh[n];
  #pragma unroll
  for (int v = 0; v < 8; ++v) {
    const int m = mrow + v + hi * 8;
    const float ht = tanhf(acc[v] + tbh);
    const float ho = hf[m * HID + n];
    const float z  = zbuf[m * HID + n];
    const float hn = (1.0f - z) * ht + z * ho;
    hf[m * HID + n] = hn;
    hb[m * HID + n] = f2bf(hn);
    if (last) hlast[m * HID + n] = hn;
  }
}

// logits for one timestep: [64,1024] x W_fc^T -> d_out[b][s][v]
__global__ void k_logits(const unsigned short* __restrict__ hb,
                         const unsigned short* __restrict__ Wfc,
                         const float* __restrict__ bfc,
                         float* __restrict__ out, int s) {
  const int lane = threadIdx.x & 31;
  const int wid  = blockIdx.x * (blockDim.x >> 5) + (threadIdx.x >> 5); // 0..31
  const int mt = wid & 3, nt = wid >> 2;          // 4 x 8 tiles
  const int mrow = mt * 16, nb = nt * 16;
  v8f acc = {};
  #pragma unroll 4
  for (int kb = 0; kb < HID / 32; ++kb) {
    const int kc = kb * 32;
    acc = wmma_bf16(load_a_frag(hb, HID, mrow, kc, lane),
                    load_b_frag(Wfc, HID, nb, kc, lane), acc);
  }
  const int l15 = lane & 15, hi = lane >> 4;
  const int n = nb + l15;
  const float bias = bfc[n];
  #pragma unroll
  for (int v = 0; v < 8; ++v) {
    const int m = mrow + v + hi * 8;
    out[(size_t)m * (SEQL * VOC) + (size_t)s * VOC + n] = acc[v] + bias;
  }
}

// ---------------- host launcher ----------------
extern "C" void kernel_launch(void* const* d_in, const int* in_sizes, int n_in,
                              void* d_out, int out_size, void* d_ws, size_t ws_size,
                              hipStream_t stream) {
  (void)in_sizes; (void)n_in; (void)out_size; (void)ws_size;
  const int*   x    = (const int*)  d_in[0];
  const float* emb  = (const float*)d_in[1];
  const float* W_r  = (const float*)d_in[2];
  const float* U_r  = (const float*)d_in[3];
  const float* b_r  = (const float*)d_in[4];
  const float* W_z  = (const float*)d_in[5];
  const float* U_z  = (const float*)d_in[6];
  const float* b_z  = (const float*)d_in[7];
  const float* W_h  = (const float*)d_in[8];
  const float* U_h  = (const float*)d_in[9];
  const float* b_h  = (const float*)d_in[10];
  const float* W_fc = (const float*)d_in[11];
  const float* b_fc = (const float*)d_in[12];
  float* out   = (float*)d_out;
  float* hlast = out + (size_t)BAT * SEQL * VOC;   // h_last tail of d_out

  // workspace carve-out (~25 MB total), 256B aligned slices
  char* ws = (char*)d_ws;
  size_t off = 0;
  auto carve = [&](size_t bytes) -> void* {
    void* p = ws + off;
    off += (bytes + 255) & ~(size_t)255;
    return p;
  };
  unsigned short* Urb  = (unsigned short*)carve((size_t)HID * HID * 2);
  unsigned short* Uzb  = (unsigned short*)carve((size_t)HID * HID * 2);
  unsigned short* Uhb  = (unsigned short*)carve((size_t)HID * HID * 2);
  unsigned short* Wrb  = (unsigned short*)carve((size_t)HID * EMB * 2);
  unsigned short* Wzb  = (unsigned short*)carve((size_t)HID * EMB * 2);
  unsigned short* Whb  = (unsigned short*)carve((size_t)HID * EMB * 2);
  unsigned short* Wfcb = (unsigned short*)carve((size_t)VOC * HID * 2);
  unsigned short* Xb   = (unsigned short*)carve((size_t)SEQL * BAT * EMB * 2);
  float*          hf   = (float*)carve((size_t)BAT * HID * 4);
  float*          zbuf = (float*)carve((size_t)BAT * HID * 4);
  unsigned short* hbb  = (unsigned short*)carve((size_t)BAT * HID * 2);
  unsigned short* rhb  = (unsigned short*)carve((size_t)BAT * HID * 2);

  // prep: bf16 weight copies (L2-resident thereafter)
  k_f32_to_bf16<<<(HID * HID) / 256, 256, 0, stream>>>(U_r, Urb, HID * HID);
  k_f32_to_bf16<<<(HID * HID) / 256, 256, 0, stream>>>(U_z, Uzb, HID * HID);
  k_f32_to_bf16<<<(HID * HID) / 256, 256, 0, stream>>>(U_h, Uhb, HID * HID);
  k_f32_to_bf16<<<(HID * EMB) / 256, 256, 0, stream>>>(W_r, Wrb, HID * EMB);
  k_f32_to_bf16<<<(HID * EMB) / 256, 256, 0, stream>>>(W_z, Wzb, HID * EMB);
  k_f32_to_bf16<<<(HID * EMB) / 256, 256, 0, stream>>>(W_h, Whb, HID * EMB);
  k_f32_to_bf16<<<(VOC * HID) / 256, 256, 0, stream>>>(W_fc, Wfcb, VOC * HID);
  k_embed<<<(SEQL * BAT * EMB) / 256, 256, 0, stream>>>(x, emb, Xb);
  k_init_h<<<(BAT * HID) / 256, 256, 0, stream>>>(hf, hbb);

  // sequential recurrence: 3 stream-ordered kernels per step
  for (int s = 0; s < SEQL; ++s) {
    k_gates<<<64, 128, 0, stream>>>(Xb, Wrb, Wzb, Urb, Uzb, hbb, hf,
                                    b_r, b_z, zbuf, rhb, s);
    k_cand<<<64, 128, 0, stream>>>(Xb, Whb, Uhb, rhb, b_h, zbuf, hf, hbb,
                                   hlast, s, (s == SEQL - 1) ? 1 : 0);
    k_logits<<<8, 128, 0, stream>>>(hbb, Wfcb, b_fc, out, s);
  }
}
